// Sequence_2061584302515
// MI455X (gfx1250) — compile-verified, matrix operand order checked by reading, not run on hardware
//
#include <hip/hip_runtime.h>
#include <hip/hip_bf16.h>

// ---------------------------------------------------------------------------
// 2-layer LSTM (H=30) over B=2048, T=1024 on gfx1250.
// One workgroup (8 wave32) per 16-row batch tile; sequential loop over T.
// Each wave owns a 16-column tile of the 120 gate outputs and holds its
// 64x16 f16 weight tile in registers; per step it issues
// 2x v_wmma_f32_16x16x32_f16 per LSTM cell (K padded 34/60 -> 64).
// Elementwise path uses v_exp_f32 + v_rcp_f32 only (no IEEE div expansion),
// with all index math hoisted out of the time loop.
// ---------------------------------------------------------------------------

#define HID    30
#define GATES  120          // 4*HID
#define TLEN   1024
#define BATCH  2048
#define DIN    4
#define DOUT   4

typedef _Float16 half_t;
typedef __attribute__((ext_vector_type(16))) _Float16 v16h;
typedef __attribute__((ext_vector_type(8)))  _Float16 v8h;
typedef __attribute__((ext_vector_type(8)))  float    v8f;

#define A_LDA  72           // halves per A-row: 64 used + 8 pad -> 144B rows,
                            // conflict-free ds_load_b128 fragment loads
#define G_LDS  132          // floats per gate-scratch row (128 cols + pad)

__device__ __forceinline__ float fast_rcp(float x) {
  return __builtin_amdgcn_rcpf(x);        // v_rcp_f32
}
__device__ __forceinline__ float sigmoid_f(float x) {
  return fast_rcp(1.0f + __expf(-x));     // v_exp_f32 + v_add + v_rcp
}
__device__ __forceinline__ float tanh_f(float x) {
  x = fminf(fmaxf(x, -15.0f), 15.0f);
  float e = __expf(2.0f * x);
  return (e - 1.0f) * fast_rcp(e + 1.0f);
}

// A-matrix fragment (16x32 f16) from LDS, per ISA 7.12.2 layout:
// lane L: row m = L&15, half-group hi = L>>4
//   halves a[0..7]  = K = kBase + 8*hi + 0..7
//   halves a[8..15] = K = kBase + 16 + 8*hi + 0..7
__device__ __forceinline__ v16h load_a_frag(const half_t* base, int kBase, int lane) {
  const int m  = lane & 15;
  const int hi = lane >> 4;
  const half_t* p = base + m * A_LDA + kBase + hi * 8;
  v8h lo  = *(const v8h*)(p);
  v8h hi8 = *(const v8h*)(p + 16);
  return __builtin_shufflevector(lo, hi8, 0,1,2,3,4,5,6,7,8,9,10,11,12,13,14,15);
}

// Combined-weight element for the padded 64xN  B-matrix (= W^T):
// cell 0: rows 0..3 = W_ih1^T, rows 4..33 = W_hh1^T, rest 0
// cell 1: rows 0..29 = W_ih2^T, rows 30..59 = W_hh2^T, rest 0
__device__ __forceinline__ half_t w_cell_elem(int cell, int n, int k,
    const float* Wih1, const float* Whh1,
    const float* Wih2, const float* Whh2) {
  if (n >= GATES) return (half_t)0.0f;
  if (cell == 0) {
    if (k < DIN)        return (half_t)Wih1[n * DIN + k];
    if (k < DIN + HID)  return (half_t)Whh1[n * HID + (k - DIN)];
  } else {
    if (k < HID)        return (half_t)Wih2[n * HID + k];
    if (k < 2 * HID)    return (half_t)Whh2[n * HID + (k - HID)];
  }
  return (half_t)0.0f;
}

// B-matrix fragment (32x16 f16), mirror of the A layout with N across lanes.
__device__ __forceinline__ v16h build_b_frag(int cell, int n, int hi, int kBase,
    const float* Wih1, const float* Whh1,
    const float* Wih2, const float* Whh2) {
  v16h b;
#pragma unroll
  for (int i = 0; i < 16; ++i) {
    int k = kBase + ((i < 8) ? (hi * 8 + i) : (16 + hi * 8 + (i - 8)));
    b[i] = w_cell_elem(cell, n, k, Wih1, Whh1, Wih2, Whh2);
  }
  return b;
}

// One LSTM elementwise update for pair offset set (precomputed indices).
__device__ __forceinline__ void lstm_ew(const float* __restrict__ gRow, int j,
                                        float* __restrict__ cArr, int cOff,
                                        float& hOut) {
  float i_ = sigmoid_f(gRow[j]);
  float f_ = sigmoid_f(gRow[HID + j]);
  float gg = tanh_f(gRow[2 * HID + j]);
  float o_ = sigmoid_f(gRow[3 * HID + j]);
  float c  = f_ * cArr[cOff] + i_ * gg;
  cArr[cOff] = c;
  hOut = o_ * tanh_f(c);
}

__global__ __launch_bounds__(256) void lstm2_wmma_kernel(
    const float* __restrict__ input,
    const float* __restrict__ Wih1, const float* __restrict__ bih1,
    const float* __restrict__ Whh1, const float* __restrict__ bhh1,
    const float* __restrict__ Wih2, const float* __restrict__ bih2,
    const float* __restrict__ Whh2, const float* __restrict__ bhh2,
    const float* __restrict__ Wlin, const float* __restrict__ blin,
    float* __restrict__ out)
{
  __shared__ __attribute__((aligned(16))) half_t A1[16 * A_LDA]; // [x | h1 | 0]
  __shared__ __attribute__((aligned(16))) half_t A2[16 * A_LDA]; // [h1 | h2 | 0]
  __shared__ float gatesLds[16 * G_LDS];                         // reused by both cells
  __shared__ float c1s[16 * HID];
  __shared__ float c2s[16 * HID];
  __shared__ float h2f[16 * 32];
  __shared__ float wlinLds[DOUT * HID];
  __shared__ float blinLds[DOUT];

  const int tid  = threadIdx.x;
  const int lane = tid & 31;
  const int wave = tid >> 5;
  const int b0   = blockIdx.x * 16;

  // ---- one-time init ----
  for (int i = tid; i < 16 * A_LDA; i += 256) { A1[i] = (half_t)0.0f; A2[i] = (half_t)0.0f; }
  for (int i = tid; i < 16 * HID;  i += 256) { c1s[i] = 0.0f; c2s[i] = 0.0f; }
  for (int i = tid; i < DOUT * HID; i += 256) wlinLds[i] = Wlin[i];
  if (tid < DOUT) blinLds[tid] = blin[tid];

  const int nCol = wave * 16 + (lane & 15);   // gate column owned by this lane
  const int hiH  = lane >> 4;

  // weight fragments stay in VGPRs for all 1024 steps
  v16h b1lo = build_b_frag(0, nCol, hiH, 0,  Wih1, Whh1, Wih2, Whh2);
  v16h b1hi = build_b_frag(0, nCol, hiH, 32, Wih1, Whh1, Wih2, Whh2);
  v16h b2lo = build_b_frag(1, nCol, hiH, 0,  Wih1, Whh1, Wih2, Whh2);
  v16h b2hi = build_b_frag(1, nCol, hiH, 32, Wih1, Whh1, Wih2, Whh2);
  const float bias1 = (nCol < GATES) ? (bih1[nCol] + bhh1[nCol]) : 0.0f;
  const float bias2 = (nCol < GATES) ? (bih2[nCol] + bhh2[nCol]) : 0.0f;

  // ---- loop-invariant elementwise indices (2 pairs max per thread) ----
  // pair 0: p = tid (< 480 always); pair 1: p = tid + 256 (valid iff tid < 224)
  const int p0  = tid;
  const int b0e = p0 / HID, j0 = p0 - b0e * HID;
  const int p1  = tid + 256;
  const bool has1 = (p1 < 16 * HID);
  const int b1e = has1 ? (p1 / HID) : 0, j1 = has1 ? (p1 - b1e * HID) : 0;
  const int g0Row = b0e * G_LDS, g1Row = b1e * G_LDS;
  const int a0Row = b0e * A_LDA, a1Row = b1e * A_LDA;
  const int h0Off = b0e * 32 + j0, h1Off = b1e * 32 + j1;

  // loop-invariant global pointers (advance 4 floats per step)
  const bool doXld  = (tid < 16 * DIN);
  const bool doProj = (tid < 16 * DOUT);
  const int  xb = tid >> 2, xk = tid & 3;          // x stage: batch row / elem
  const float* xPtr = input + (size_t)(b0 + xb) * TLEN * DIN + xk;
  float*       oPtr = out   + (size_t)(b0 + xb) * TLEN * DOUT + xk;
  const float* pfPtr = input + (size_t)(b0 + (tid & 15)) * TLEN * DIN;

  __syncthreads();

  for (int t = 0; t < TLEN; ++t) {
    // (a) stage x_t into A1 columns [0, DIN)
    if (doXld) A1[xb * A_LDA + xk] = (half_t)xPtr[(size_t)t * DIN];
    if (tid < 16) {       // stream-ahead prefetch -> global_prefetch_b8
      int tpf = (t + 24 < TLEN) ? (t + 24) : t;
      __builtin_prefetch(pfPtr + (size_t)tpf * DIN, 0, 0);
    }
    __syncthreads();

    // (b) cell-1 GEMM: gates1 = [x | h1] @ W1^T + bias1
    {
      v8f acc = {bias1, bias1, bias1, bias1, bias1, bias1, bias1, bias1};
      v16h a0 = load_a_frag(A1, 0,  lane);
      v16h a1 = load_a_frag(A1, 32, lane);
      acc = __builtin_amdgcn_wmma_f32_16x16x32_f16(false, a0, false, b1lo, (short)0, acc, false, false);
      acc = __builtin_amdgcn_wmma_f32_16x16x32_f16(false, a1, false, b1hi, (short)0, acc, false, false);
#pragma unroll
      for (int r = 0; r < 8; ++r)
        gatesLds[(r + 8 * hiH) * G_LDS + nCol] = acc[r];
    }
    __syncthreads();

    // (c) elementwise cell-1 update: h1 -> A1 (next step) and A2 (this step)
    {
      float h;
      lstm_ew(&gatesLds[g0Row], j0, c1s, p0, h);
      A1[a0Row + DIN + j0] = (half_t)h;
      A2[a0Row + j0]       = (half_t)h;
      if (has1) {
        lstm_ew(&gatesLds[g1Row], j1, c1s, p1, h);
        A1[a1Row + DIN + j1] = (half_t)h;
        A2[a1Row + j1]       = (half_t)h;
      }
    }
    __syncthreads();

    // (d) cell-2 GEMM: gates2 = [h1 | h2] @ W2^T + bias2
    {
      v8f acc = {bias2, bias2, bias2, bias2, bias2, bias2, bias2, bias2};
      v16h a0 = load_a_frag(A2, 0,  lane);
      v16h a1 = load_a_frag(A2, 32, lane);
      acc = __builtin_amdgcn_wmma_f32_16x16x32_f16(false, a0, false, b2lo, (short)0, acc, false, false);
      acc = __builtin_amdgcn_wmma_f32_16x16x32_f16(false, a1, false, b2hi, (short)0, acc, false, false);
#pragma unroll
      for (int r = 0; r < 8; ++r)
        gatesLds[(r + 8 * hiH) * G_LDS + nCol] = acc[r];
    }
    __syncthreads();

    // (e) elementwise cell-2 update: h2 -> A2 (next step) and h2f (projection)
    {
      float h;
      lstm_ew(&gatesLds[g0Row], j0, c2s, p0, h);
      A2[a0Row + HID + j0] = (half_t)h;
      h2f[h0Off] = h;
      if (has1) {
        lstm_ew(&gatesLds[g1Row], j1, c2s, p1, h);
        A2[a1Row + HID + j1] = (half_t)h;
        h2f[h1Off] = h;
      }
    }
    __syncthreads();

    // (f) output projection: out[b,t,:] = h2 @ Wlin^T + blin (tiny -> VALU)
    if (doProj) {
      float acc = blinLds[xk];
#pragma unroll
      for (int j = 0; j < HID; ++j)
        acc += h2f[xb * 32 + j] * wlinLds[xk * HID + j];
      oPtr[(size_t)t * DOUT] = acc;
    }
    // no barrier needed: next writers of h2f/gatesLds are behind >=1 barrier
  }
}

extern "C" void kernel_launch(void* const* d_in, const int* in_sizes, int n_in,
                              void* d_out, int out_size, void* d_ws, size_t ws_size,
                              hipStream_t stream) {
  const float* input = (const float*)d_in[0];
  const float* Wih1  = (const float*)d_in[1];
  const float* bih1  = (const float*)d_in[2];
  const float* Whh1  = (const float*)d_in[3];
  const float* bhh1  = (const float*)d_in[4];
  const float* Wih2  = (const float*)d_in[5];
  const float* bih2  = (const float*)d_in[6];
  const float* Whh2  = (const float*)d_in[7];
  const float* bhh2  = (const float*)d_in[8];
  const float* Wlin  = (const float*)d_in[9];
  const float* blin  = (const float*)d_in[10];
  float* out = (float*)d_out;

  dim3 grid(BATCH / 16);   // 128 workgroups, one 16-row batch tile each
  dim3 block(256);         // 8 wave32
  lstm2_wmma_kernel<<<grid, block, 0, stream>>>(
      input, Wih1, bih1, Whh1, bhh1, Wih2, bih2, Whh2, bhh2, Wlin, blin, out);
}
